// MambaBackbone_7370163880637
// MI455X (gfx1250) — matrix operand order repaired and checked by reading, not compiled, not run on hardware
//
#include <hip/hip_runtime.h>
#include <hip/hip_bf16.h>

// ---------------------------------------------------------------------------
// Mamba backbone for MI455X (gfx1250): bf16 WMMA GEMMs + wave32 selective scan
// GEMM: 128x64 workgroup tile, 32x32 per-wave tile (4 WMMA/K-step),
//       double-buffered LDS, single barrier per K-step.
// ---------------------------------------------------------------------------

typedef __attribute__((ext_vector_type(16))) __bf16 v16bf;
typedef __attribute__((ext_vector_type(8)))  float  v8f;

#define D_MODEL 1024
#define D_STATE 16
#define D_CONV  4
#define D_INNER 2048
#define DT_RANK 64
#define B_SZ    2
#define SEQ     2048
#define DEPTH   8
#define ROWS    (B_SZ * SEQ)             // 4096
#define XDBL_W  (DT_RANK + 2 * D_STATE)  // 96

// ---- helpers --------------------------------------------------------------

__device__ __forceinline__ unsigned short f2bf_rne(float f) {
  unsigned u = __float_as_uint(f);
  if ((u & 0x7F800000u) == 0x7F800000u) return (unsigned short)(u >> 16); // inf/nan
  unsigned r = u + 0x7FFFu + ((u >> 16) & 1u);
  return (unsigned short)(r >> 16);
}

__device__ __forceinline__ float siluf(float v) {
  return v / (1.f + __expf(-v));
}

__device__ __forceinline__ float softplusf(float x) {
  return (x > 20.f) ? x : log1pf(__expf(x));
}

// ---- elementwise kernels --------------------------------------------------

__global__ void cvt_f32_bf16_kernel(const float* __restrict__ src,
                                    unsigned short* __restrict__ dst, long n) {
  long i = (long)blockIdx.x * blockDim.x + threadIdx.x;
  if (i < n) dst[i] = f2bf_rne(src[i]);
}

// causal depthwise conv (K=4) + bias + SiLU; reads first D_INNER cols of xz.
// Writes fp32 (for the scan) and bf16 (for the W_x GEMM) in one pass.
__global__ void conv_silu_kernel(const float* __restrict__ xz,
                                 const float* __restrict__ cw,
                                 const float* __restrict__ cb,
                                 float* __restrict__ out_f32,
                                 unsigned short* __restrict__ out_bf,
                                 long total) {
  long i = (long)blockIdx.x * blockDim.x + threadIdx.x;
  if (i >= total) return;
  int  c  = (int)(i % D_INNER);
  long bl = i / D_INNER;              // b*SEQ + l
  int  l  = (int)(bl % SEQ);
  float acc = cb[c];
#pragma unroll
  for (int k = 0; k < D_CONV; ++k) {
    int ls = l - (D_CONV - 1) + k;
    if (ls >= 0)
      acc += xz[(bl - l + ls) * (2 * D_INNER) + c] * cw[c * D_CONV + k];
  }
  float v = siluf(acc);
  out_f32[i] = v;
  out_bf[i]  = f2bf_rne(v);
}

// slice dt_raw (cols 0..63 of x_dbl) -> bf16
__global__ void dtslice_kernel(const float* __restrict__ xdbl,
                               unsigned short* __restrict__ dst, long total) {
  long i = (long)blockIdx.x * blockDim.x + threadIdx.x;
  if (i >= total) return;
  int  r   = (int)(i & (DT_RANK - 1));
  long row = i >> 6;
  dst[i] = f2bf_rne(xdbl[row * XDBL_W + r]);
}

// selective scan: 16 lanes per (b, c) channel, one lane per SSM state
__global__ __launch_bounds__(256)
void scan_kernel(const float* __restrict__ delta, const float* __restrict__ xdbl,
                 const float* __restrict__ u,     const float* __restrict__ Alog,
                 const float* __restrict__ Dp,    float* __restrict__ y) {
  int gid = blockIdx.x * blockDim.x + threadIdx.x;     // over B*C*16
  int n   = gid & (D_STATE - 1);
  int ch  = gid >> 4;                                  // b*C + c
  int c   = ch % D_INNER;
  int b   = ch / D_INNER;
  float A  = -__expf(Alog[c * D_STATE + n]);
  float Dv = Dp[c];
  float h  = 0.f;
  size_t base = (size_t)b * SEQ;
  for (int l = 0; l < SEQ; ++l) {
    size_t row = base + l;
    float d  = delta[row * D_INNER + c];
    float uv = u[row * D_INNER + c];
    float Bv = xdbl[row * XDBL_W + DT_RANK + n];
    float Cv = xdbl[row * XDBL_W + DT_RANK + D_STATE + n];
    h = h * __expf(d * A) + d * uv * Bv;
    float p = h * Cv;
    p += __shfl_xor(p, 1, 32);
    p += __shfl_xor(p, 2, 32);
    p += __shfl_xor(p, 4, 32);
    p += __shfl_xor(p, 8, 32);
    if (n == 0) y[row * D_INNER + c] = p + uv * Dv;
  }
}

// y * silu(z) -> bf16 (z = second half of xz)
__global__ void gate_kernel(const float* __restrict__ yss,
                            const float* __restrict__ xz,
                            unsigned short* __restrict__ out, long total) {
  long i = (long)blockIdx.x * blockDim.x + threadIdx.x;
  if (i >= total) return;
  int  c   = (int)(i % D_INNER);
  long row = i / D_INNER;
  float z = xz[row * (2 * D_INNER) + D_INNER + c];
  out[i] = f2bf_rne(yss[i] * siluf(z));
}

// nan_to_num copy to output
__global__ void finalize_kernel(const float* __restrict__ src,
                                float* __restrict__ dst, long n) {
  long i = (long)blockIdx.x * blockDim.x + threadIdx.x;
  if (i >= n) return;
  float v = src[i];
  if (v != v) v = 0.f;
  else if (v >  3.402823466e38f) v =  3.402823466e38f;
  else if (v < -3.402823466e38f) v = -3.402823466e38f;
  dst[i] = v;
}

// ---- bf16 WMMA GEMM: C[M,N] (+)= A[M,K] * B[N,K]^T ------------------------
// epilogue 0: store; 1: store softplus(acc + bias[n]); 2: C += acc
#define GBM 128
#define GBN 64
#define GKT 32

union AFragU { unsigned u[8]; v16bf v; };

__global__ __launch_bounds__(256, 2)
void gemm_bf16_wmma_kernel(const unsigned short* __restrict__ A,
                           const unsigned short* __restrict__ B,
                           float* __restrict__ C,
                           int M, int N, int K,
                           int epilogue, const float* __restrict__ bias) {
  __shared__ unsigned short sA[2][GBM * GKT];   // 2 x 8 KB
  __shared__ unsigned short sB[2][GBN * GKT];   // 2 x 4 KB

  const int tid  = threadIdx.x;
  const int lane = tid & 31;
  const int wave = tid >> 5;       // 0..7
  const int wm   = wave >> 1;      // 0..3 : 32-row slab
  const int wn   = wave & 1;       // 0..1 : 32-col half
  const int gm   = blockIdx.y * GBM;
  const int gn   = blockIdx.x * GBN;

  // staging: A = 128x32 bf16 = 512 uint4 (2/thread), B = 64x32 = 256 uint4
  const int ar0 = tid >> 2;              // rows 0..63
  const int ar1 = ar0 + 64;              // rows 64..127
  const int ac  = (tid & 3) * 8;         // col element offset 0,8,16,24
  const int br  = tid >> 2;              // 0..63
  const int bc  = (tid & 3) * 8;

  v8f acc00 = {}; v8f acc01 = {}; v8f acc10 = {}; v8f acc11 = {};

  // ---- prologue: stage K-tile 0 into buffer 0 ----
  {
    uint4 ra0 = *(const uint4*)(A + (size_t)(gm + ar0) * K + ac);
    uint4 ra1 = *(const uint4*)(A + (size_t)(gm + ar1) * K + ac);
    uint4 rb  = {0u, 0u, 0u, 0u};
    int bn = gn + br;
    if (bn < N) rb = *(const uint4*)(B + (size_t)bn * K + bc);
    *(uint4*)&sA[0][ar0 * GKT + ac] = ra0;
    *(uint4*)&sA[0][ar1 * GKT + ac] = ra1;
    *(uint4*)&sB[0][br * GKT + bc]  = rb;
  }
  __syncthreads();

  int p = 0;
  for (int k0 = 0; k0 < K; k0 += GKT) {
    const bool has_next = (k0 + GKT) < K;

    // issue next tile's global loads early (latency overlaps WMMA below)
    uint4 ra0, ra1, rb;
    if (has_next) {
      const int kn = k0 + GKT;
      const unsigned short* pa0 = A + (size_t)(gm + ar0) * K + kn + ac;
      ra0 = *(const uint4*)pa0;
      ra1 = *(const uint4*)(A + (size_t)(gm + ar1) * K + kn + ac);
      rb = make_uint4(0u, 0u, 0u, 0u);
      int bn = gn + br;
      if (bn < N) rb = *(const uint4*)(B + (size_t)bn * K + kn + bc);
      if (kn + GKT < K) __builtin_prefetch(pa0 + GKT, 0, 1);  // warm L2 two tiles out
    }

    // ---- fragments from buffer p (CDNA5 16-bit WMMA VGPR layouts) ----
    AFragU a0, a1, b0, b1;
    {
      const int mr0 = wm * 32 + (lane & 15);
      const int mr1 = mr0 + 16;
      const int akb = (lane < 16) ? 0 : 8;
#pragma unroll
      for (int v = 0; v < 4; ++v) {
        a0.u[v]     = *(const unsigned*)&sA[p][mr0 * GKT + akb + 2 * v];
        a0.u[4 + v] = *(const unsigned*)&sA[p][mr0 * GKT + akb + 16 + 2 * v];
        a1.u[v]     = *(const unsigned*)&sA[p][mr1 * GKT + akb + 2 * v];
        a1.u[4 + v] = *(const unsigned*)&sA[p][mr1 * GKT + akb + 16 + 2 * v];
      }
      const int nb0 = wn * 32 + (lane & 15);
      const int nb1 = nb0 + 16;
      const int bkb = (lane < 16) ? 0 : 8;
#pragma unroll
      for (int v = 0; v < 8; ++v) {
        b0.u[v] = *(const unsigned*)&sB[p][nb0 * GKT + bkb + 2 * v + ((v >= 4) ? 8 : 0)];
        b1.u[v] = *(const unsigned*)&sB[p][nb1 * GKT + bkb + 2 * v + ((v >= 4) ? 8 : 0)];
      }
    }

    acc00 = __builtin_amdgcn_wmma_f32_16x16x32_bf16(false, a0.v, false, b0.v,
                                                    (short)0, acc00, false, false);
    acc01 = __builtin_amdgcn_wmma_f32_16x16x32_bf16(false, a0.v, false, b1.v,
                                                    (short)0, acc01, false, false);
    acc10 = __builtin_amdgcn_wmma_f32_16x16x32_bf16(false, a1.v, false, b0.v,
                                                    (short)0, acc10, false, false);
    acc11 = __builtin_amdgcn_wmma_f32_16x16x32_bf16(false, a1.v, false, b1.v,
                                                    (short)0, acc11, false, false);

    if (has_next) {
      const int q = p ^ 1;
      *(uint4*)&sA[q][ar0 * GKT + ac] = ra0;
      *(uint4*)&sA[q][ar1 * GKT + ac] = ra1;
      *(uint4*)&sB[q][br * GKT + bc]  = rb;
      __syncthreads();          // single barrier per K-step
      p = q;
    }
  }

  // ---- epilogue: CDNA5 16x16 f32 C layout (VGPR v -> M=v / v+8) ----
  const int nc0 = gn + wn * 32 + (lane & 15);
  const int nc1 = nc0 + 16;
  const int mb0 = gm + wm * 32 + ((lane < 16) ? 0 : 8);
  const int mb1 = mb0 + 16;
#pragma unroll
  for (int v = 0; v < 8; ++v) {
    float e00 = acc00[v], e01 = acc01[v], e10 = acc10[v], e11 = acc11[v];
    if (epilogue == 1) {
      if (nc0 < N) { float bz = bias[nc0]; e00 = softplusf(e00 + bz); e10 = softplusf(e10 + bz); }
      if (nc1 < N) { float bz = bias[nc1]; e01 = softplusf(e01 + bz); e11 = softplusf(e11 + bz); }
    }
    const int m0 = mb0 + v, m1 = mb1 + v;
    if (nc0 < N) {
      size_t i00 = (size_t)m0 * N + nc0, i10 = (size_t)m1 * N + nc0;
      if (epilogue == 2) { C[i00] += e00; C[i10] += e10; }
      else               { C[i00]  = e00; C[i10]  = e10; }
    }
    if (nc1 < N) {
      size_t i01 = (size_t)m0 * N + nc1, i11 = (size_t)m1 * N + nc1;
      if (epilogue == 2) { C[i01] += e01; C[i11] += e11; }
      else               { C[i01]  = e01; C[i11]  = e11; }
    }
  }
}

// ---------------------------------------------------------------------------

extern "C" void kernel_launch(void* const* d_in, const int* in_sizes, int n_in,
                              void* d_out, int out_size, void* d_ws, size_t ws_size,
                              hipStream_t stream) {
  const float* x      = (const float*)d_in[0];
  const float* W_in   = (const float*)d_in[1];
  const float* conv_w = (const float*)d_in[2];
  const float* conv_b = (const float*)d_in[3];
  const float* W_x    = (const float*)d_in[4];
  const float* W_dt   = (const float*)d_in[5];
  const float* b_dt   = (const float*)d_in[6];
  const float* A_log  = (const float*)d_in[7];
  const float* D_par  = (const float*)d_in[8];
  const float* W_out  = (const float*)d_in[9];

  char* ws = (char*)d_ws;
  size_t off = 0;
  auto take = [&](size_t bytes) {
    size_t o = off;
    off += (bytes + 255) & ~(size_t)255;
    return o;
  };

  float*          xcur   = (float*)(ws + take((size_t)ROWS * D_MODEL * 4));
  unsigned short* actbf  = (unsigned short*)(ws + take((size_t)ROWS * D_INNER * 2));
  unsigned short* wbf_in = (unsigned short*)(ws + take((size_t)2 * D_INNER * D_MODEL * 2));
  unsigned short* wbf_x  = (unsigned short*)(ws + take((size_t)XDBL_W * D_INNER * 2));
  unsigned short* wbf_dt = (unsigned short*)(ws + take((size_t)D_INNER * DT_RANK * 2));
  unsigned short* wbf_o  = (unsigned short*)(ws + take((size_t)D_MODEL * D_INNER * 2));
  float*          xz     = (float*)(ws + take((size_t)ROWS * 2 * D_INNER * 4));
  float*          xinp   = (float*)(ws + take((size_t)ROWS * D_INNER * 4));
  float*          xdbl   = (float*)(ws + take((size_t)ROWS * XDBL_W * 4));
  unsigned short* dtrbf  = (unsigned short*)(ws + take((size_t)ROWS * DT_RANK * 2));
  float*          delta  = (float*)(ws + take((size_t)ROWS * D_INNER * 4));
  float*          yss    = (float*)(ws + take((size_t)ROWS * D_INNER * 4));

  auto blk1d = [](long n) { return (unsigned)((n + 255) / 256); };

  hipMemcpyAsync(xcur, x, (size_t)ROWS * D_MODEL * sizeof(float),
                 hipMemcpyDeviceToDevice, stream);

  for (int layer = 0; layer < DEPTH; ++layer) {
    const long n_win = (long)2 * D_INNER * D_MODEL;
    const long n_wx  = (long)XDBL_W * D_INNER;
    const long n_wdt = (long)D_INNER * DT_RANK;
    const long n_wo  = (long)D_MODEL * D_INNER;

    cvt_f32_bf16_kernel<<<blk1d(n_win), 256, 0, stream>>>(
        W_in + (size_t)layer * n_win, wbf_in, n_win);
    cvt_f32_bf16_kernel<<<blk1d(n_wx), 256, 0, stream>>>(
        W_x + (size_t)layer * n_wx, wbf_x, n_wx);
    cvt_f32_bf16_kernel<<<blk1d(n_wdt), 256, 0, stream>>>(
        W_dt + (size_t)layer * n_wdt, wbf_dt, n_wdt);
    cvt_f32_bf16_kernel<<<blk1d(n_wo), 256, 0, stream>>>(
        W_out + (size_t)layer * n_wo, wbf_o, n_wo);

    // x -> bf16
    cvt_f32_bf16_kernel<<<blk1d((long)ROWS * D_MODEL), 256, 0, stream>>>(
        xcur, actbf, (long)ROWS * D_MODEL);

    // GEMM1: xz = x @ W_in^T   [4096 x 4096 x 1024]
    gemm_bf16_wmma_kernel<<<dim3((2 * D_INNER) / GBN, ROWS / GBM), 256, 0, stream>>>(
        actbf, wbf_in, xz, ROWS, 2 * D_INNER, D_MODEL, 0, nullptr);

    // causal conv + SiLU (writes fp32 for scan + bf16 for GEMM2)
    conv_silu_kernel<<<blk1d((long)ROWS * D_INNER), 256, 0, stream>>>(
        xz, conv_w + (size_t)layer * D_INNER * D_CONV,
        conv_b + (size_t)layer * D_INNER, xinp, actbf, (long)ROWS * D_INNER);

    // GEMM2: x_dbl = xin @ W_x^T   [4096 x 96 x 2048]
    gemm_bf16_wmma_kernel<<<dim3((XDBL_W + GBN - 1) / GBN, ROWS / GBM), 256, 0, stream>>>(
        actbf, wbf_x, xdbl, ROWS, XDBL_W, D_INNER, 0, nullptr);

    // dt_raw slice -> bf16
    dtslice_kernel<<<blk1d((long)ROWS * DT_RANK), 256, 0, stream>>>(
        xdbl, dtrbf, (long)ROWS * DT_RANK);

    // GEMM3: delta = softplus(dt_raw @ W_dt^T + b_dt)   [4096 x 2048 x 64]
    gemm_bf16_wmma_kernel<<<dim3(D_INNER / GBN, ROWS / GBM), 256, 0, stream>>>(
        dtrbf, wbf_dt, delta, ROWS, D_INNER, DT_RANK, 1,
        b_dt + (size_t)layer * D_INNER);

    // selective scan
    scan_kernel<<<(B_SZ * D_INNER * D_STATE) / 256, 256, 0, stream>>>(
        delta, xdbl, xinp,
        A_log + (size_t)layer * D_INNER * D_STATE,
        D_par + (size_t)layer * D_INNER, yss);

    // gate: y * silu(z) -> bf16
    gate_kernel<<<blk1d((long)ROWS * D_INNER), 256, 0, stream>>>(
        yss, xz, actbf, (long)ROWS * D_INNER);

    // GEMM4: xcur += y @ W_out^T   [4096 x 1024 x 2048]
    gemm_bf16_wmma_kernel<<<dim3(D_MODEL / GBN, ROWS / GBM), 256, 0, stream>>>(
        actbf, wbf_o, xcur, ROWS, D_MODEL, D_INNER, 2, nullptr);
  }

  finalize_kernel<<<blk1d((long)ROWS * D_MODEL), 256, 0, stream>>>(
      xcur, (float*)d_out, (long)ROWS * D_MODEL);
}
